// VectorQuantizer_31834297598606
// MI455X (gfx1250) — matrix operand order repaired and checked by reading, not compiled, not run on hardware
//
#include <hip/hip_runtime.h>

// ---------------------------------------------------------------------------
// VQ argmin for MI455X (gfx1250): fused GEMM (bf16x3 emulated fp32) + argmin.
//   dist2 = |c|^2 - 2 x.c   (|x|^2 dropped: constant per row)
//   x.c  ~= x_hi.c_hi + x_hi.c_lo + x_lo.c_hi   (bf16 WMMA, f32 accumulate)
// Codebook tiles are DMA'd into LDS with GLOBAL_LOAD_ASYNC_TO_LDS_B128
// (ASYNCcnt) and double-buffered so the transfer overlaps the WMMA work.
// ---------------------------------------------------------------------------

typedef __attribute__((ext_vector_type(16))) __bf16 v16bf;
typedef __attribute__((ext_vector_type(8)))  __bf16 v8bf;
typedef __attribute__((ext_vector_type(8)))  float  v8f;

union Frag {
  v16bf v;
  v8bf  h[2];
};

#define DIM            256
#define KSTEPS         8        // DIM / 32
#define NTILE          16       // codes per N tile
#define ROWS_PER_WAVE  16
#define WAVES_PER_BLK  8
#define ROWS_PER_BLK   (ROWS_PER_WAVE * WAVES_PER_BLK)   // 128

// ---------------------------------------------------------------------------
// Async global->LDS 16B DMA (CDNA5, tracked by ASYNCcnt). The LDS address is
// the low 32 bits of the generic shared-memory pointer (flat LDS aperture:
// LDS offset in the low 32 bits).
// ---------------------------------------------------------------------------
__device__ __forceinline__ void async_ld_b128(const void* gsrc, void* ldst) {
  const unsigned ldsa = (unsigned)(size_t)ldst;
  asm volatile("global_load_async_to_lds_b128 %0, %1, off"
               :: "v"(ldsa), "v"(gsrc)
               : "memory");
}

// ---------------------------------------------------------------------------
// Prepass: split codebook into bf16 hi/lo planes; compute |c|^2 in fp32.
// One block (256 threads) per code row.
// ---------------------------------------------------------------------------
__global__ void vq_prepass(const float* __restrict__ cb,
                           __bf16* __restrict__ cb_hi,
                           __bf16* __restrict__ cb_lo,
                           float* __restrict__ c2) {
  const int k = blockIdx.x;
  const int t = threadIdx.x;              // 0..255 == DIM
  const float f  = cb[(size_t)k * DIM + t];
  const __bf16 h = (__bf16)f;
  cb_hi[(size_t)k * DIM + t] = h;
  cb_lo[(size_t)k * DIM + t] = (__bf16)(f - (float)h);

  float s = f * f;
  #pragma unroll
  for (int off = 16; off > 0; off >>= 1)
    s += __shfl_xor(s, off, 32);

  __shared__ float red[WAVES_PER_BLK];
  if ((t & 31) == 0) red[t >> 5] = s;
  __syncthreads();
  if (t == 0) {
    float tot = 0.f;
    #pragma unroll
    for (int w = 0; w < WAVES_PER_BLK; ++w) tot += red[w];
    c2[k] = tot;
  }
}

// ---------------------------------------------------------------------------
// Main kernel: each wave owns a 16-row M tile (A fragments register-resident
// across the entire K loop). All 8 waves share double-buffered LDS codebook
// tiles fed by async DMA.
// ---------------------------------------------------------------------------
__global__ void __launch_bounds__(256, 1)
vq_argmin(const float* __restrict__ x,
          const __bf16* __restrict__ cb_hi,
          const __bf16* __restrict__ cb_lo,
          const float* __restrict__ c2,
          int* __restrict__ out,
          int K) {
  const int tid  = threadIdx.x;
  const int lane = tid & 31;
  const int wave = tid >> 5;
  const int half = lane >> 4;          // 0: lanes 0-15, 1: lanes 16-31
  const int r    = lane & 15;

  const int rowBase = blockIdx.x * ROWS_PER_BLK + wave * ROWS_PER_WAVE;
  const size_t myRow = (size_t)(rowBase + r);

  // ---- Build A fragments (hi/lo) straight from global fp32 ----------------
  // A operand, this code's k order (same as B): lane half h holds contiguous K [16h,16h+16)
  Frag a_hi[KSTEPS], a_lo[KSTEPS];
  {
    const float* xr = x + myRow * DIM;
    #pragma unroll
    for (int s = 0; s < KSTEPS; ++s) {
      #pragma unroll
      for (int c = 0; c < 2; ++c) {
        const int k0 = 32 * s + 16 * half + 8 * c;
        const float4 f0 = *(const float4*)(xr + k0);
        const float4 f1 = *(const float4*)(xr + k0 + 4);
        const float tmp[8] = {f0.x, f0.y, f0.z, f0.w, f1.x, f1.y, f1.z, f1.w};
        #pragma unroll
        for (int e = 0; e < 8; ++e) {
          const __bf16 hh = (__bf16)tmp[e];
          a_hi[s].h[c][e] = hh;
          a_lo[s].h[c][e] = (__bf16)(tmp[e] - (float)hh);
        }
      }
    }
  }

  // Double-buffered codebook tile: [buf][NTILE codes][DIM], hi + lo planes.
  __shared__ __align__(16) __bf16 lds_hi[2][NTILE * DIM];   // 2 x 8 KB
  __shared__ __align__(16) __bf16 lds_lo[2][NTILE * DIM];   // 2 x 8 KB

  // Issue the 4 async 16B chunks this thread owns for tile `t` into `buf`.
  auto issue_tile = [&](int t, int buf) {
    const uint4* sh = (const uint4*)(cb_hi + (size_t)t * NTILE * DIM);
    const uint4* sl = (const uint4*)(cb_lo + (size_t)t * NTILE * DIM);
    uint4* dh = (uint4*)lds_hi[buf];
    uint4* dl = (uint4*)lds_lo[buf];
    async_ld_b128(sh + tid,       dh + tid);
    async_ld_b128(sh + tid + 256, dh + tid + 256);
    async_ld_b128(sl + tid,       dl + tid);
    async_ld_b128(sl + tid + 256, dl + tid + 256);
  };

  float bv[8];
  int   bi[8];
  #pragma unroll
  for (int g = 0; g < 8; ++g) { bv[g] = __builtin_inff(); bi[g] = 0; }

  const int ntiles = K / NTILE;   // 64

  issue_tile(0, 0);               // prologue: tile 0 in flight

  for (int nt = 0; nt < ntiles; ++nt) {
    const int buf = nt & 1;
    if (nt + 1 < ntiles) {
      // Overlap: start DMA of tile nt+1, then wait only for tile nt's 4
      // chunks (async loads complete in order; the next tile's 4 may remain
      // outstanding while we compute).
      issue_tile(nt + 1, buf ^ 1);
      asm volatile("s_wait_asynccnt 0x4" ::: "memory");
    } else {
      asm volatile("s_wait_asynccnt 0x0" ::: "memory");
    }
    __syncthreads();   // publish tile nt's LDS data to all 8 waves

    v8f acc = {0.f, 0.f, 0.f, 0.f, 0.f, 0.f, 0.f, 0.f};
    #pragma unroll
    for (int s = 0; s < KSTEPS; ++s) {
      // B operand, this code's k order (same as A): lane (half h, col r) holds contiguous
      // K = [32s + 16h, 32s + 16h + 16) of code r  -> one 32B LDS read.
      const int kb = 32 * s + 16 * half;
      Frag b_hi, b_lo;
      b_hi.v = *(const v16bf*)(lds_hi[buf] + r * DIM + kb);
      b_lo.v = *(const v16bf*)(lds_lo[buf] + r * DIM + kb);

      acc = __builtin_amdgcn_wmma_f32_16x16x32_bf16(
          false, a_hi[s].v, false, b_hi.v, (short)0, acc, false, false);
      acc = __builtin_amdgcn_wmma_f32_16x16x32_bf16(
          false, a_hi[s].v, false, b_lo.v, (short)0, acc, false, false);
      acc = __builtin_amdgcn_wmma_f32_16x16x32_bf16(
          false, a_lo[s].v, false, b_hi.v, (short)0, acc, false, false);
    }

    // C/D layout: lane (half h, n=r), VGPR g -> row rowBase + 8h + g, col n.
    const int n  = nt * NTILE + r;
    const float cc = c2[n];
    #pragma unroll
    for (int g = 0; g < 8; ++g) {
      const float score = cc - 2.0f * acc[g];
      if (score < bv[g]) { bv[g] = score; bi[g] = n; }  // strict < keeps first
    }

    // All waves done reading buf before iteration nt+1 DMAs into buf^1's
    // partner (which is this buf again at nt+2) -- and before the next
    // issue targets buf^1 at nt+1 (that buffer was last read at nt-1).
    __syncthreads();
  }

  // Cross-lane argmin: the 16 lanes of each half share the same 8 rows.
  #pragma unroll
  for (int off = 1; off < 16; off <<= 1) {
    #pragma unroll
    for (int g = 0; g < 8; ++g) {
      const float ov = __shfl_xor(bv[g], off, 32);
      const int   oi = __shfl_xor(bi[g], off, 32);
      if (ov < bv[g] || (ov == bv[g] && oi < bi[g])) { bv[g] = ov; bi[g] = oi; }
    }
  }
  if (r == 0) {
    #pragma unroll
    for (int g = 0; g < 8; ++g)
      out[rowBase + 8 * half + g] = bi[g];
  }
}

// ---------------------------------------------------------------------------
// Launch
// ---------------------------------------------------------------------------
extern "C" void kernel_launch(void* const* d_in, const int* in_sizes, int n_in,
                              void* d_out, int out_size, void* d_ws, size_t ws_size,
                              hipStream_t stream) {
  const float* x  = (const float*)d_in[0];   // (B,T,D) fp32, B*T = rows
  const float* cb = (const float*)d_in[1];   // (K,D)   fp32

  const int rows = in_sizes[0] / DIM;        // 32768
  const int K    = in_sizes[1] / DIM;        // 1024

  // Workspace layout: cb_hi (K*D bf16) | cb_lo (K*D bf16) | c2 (K f32)
  __bf16* cb_hi = (__bf16*)d_ws;
  __bf16* cb_lo = cb_hi + (size_t)K * DIM;
  float*  c2    = (float*)(cb_lo + (size_t)K * DIM);

  vq_prepass<<<K, 256, 0, stream>>>(cb, cb_hi, cb_lo, c2);
  vq_argmin<<<rows / ROWS_PER_BLK, 256, 0, stream>>>(
      x, cb_hi, cb_lo, c2, (int*)d_out, K);
}
